// MockSelfAttention_81381040325143
// MI455X (gfx1250) — matrix operand order repaired and checked by reading, not compiled, not run on hardware
//
#include <hip/hip_runtime.h>

// ---------------- problem constants ----------------
#define Bc 4
#define Sc 2048
#define Hc 1024
#define BSc (Bc * Sc) // 8192 total rows

typedef unsigned short u16;
typedef __attribute__((ext_vector_type(16))) __bf16 v16bf;
typedef __attribute__((ext_vector_type(8)))  float  v8f;

union Frag {
    v16bf bf;
    uint4 q[2];
    u16   u[16];
};

__device__ inline u16 f2bf(float f) {
    unsigned int u = __float_as_uint(f);
    u += 0x7FFFu + ((u >> 16) & 1u);   // round-to-nearest-even
    return (u16)(u >> 16);
}
__device__ inline float bf2f(u16 h) {
    return __uint_as_float(((unsigned int)h) << 16);
}

// ---------------- fp32 -> bf16 conversion ----------------
__global__ __launch_bounds__(256) void cvt_f32_bf16(const float* __restrict__ src,
                                                    u16* __restrict__ dst, int n) {
    int i = blockIdx.x * blockDim.x + threadIdx.x;
    int stride = gridDim.x * blockDim.x;
    for (; i < n; i += stride) dst[i] = f2bf(src[i]);
}

// ---------------- bf16 32x32 tiled transpose: dst[b][h][s] = src[b][s][h] -------------
__global__ __launch_bounds__(256) void transpose_bf16(const u16* __restrict__ src,  // [B*S,H]
                                                      u16* __restrict__ dst) {      // [B*H,S]
    __shared__ u16 tile[32][33];
    const int b  = blockIdx.z;
    const int s0 = blockIdx.y * 32;
    const int h0 = blockIdx.x * 32;
    const int tx = threadIdx.x;   // 0..31
    const int ty = threadIdx.y;   // 0..7
#pragma unroll
    for (int i = 0; i < 32; i += 8)
        tile[ty + i][tx] = src[((size_t)b * Sc + s0 + ty + i) * Hc + h0 + tx];
    __syncthreads();
#pragma unroll
    for (int i = 0; i < 32; i += 8)
        dst[((size_t)b * Hc + h0 + ty + i) * Sc + s0 + tx] = tile[tx][ty + i];
}

// ---------------- LDS-tiled bf16 WMMA GEMM:  Y = A @ W^T + bias ----------------
// A: [M,K] bf16 row-major, W: [N,K] bf16 row-major (torch Linear weight)
// 128x128 block tile, 8 waves, each wave: 32x64 (2x4 grid of 16x16x32 WMMA)
__global__ __launch_bounds__(256) void gemm_bf16_wmma(
    const u16* __restrict__ A, const u16* __restrict__ W,
    const float* __restrict__ bias,
    u16* __restrict__ outB, float* __restrict__ outF,
    int M, int N, int K, int writeF32)
{
    __shared__ __align__(16) u16 As[128 * 32];
    __shared__ __align__(16) u16 Ws[128 * 32];

    const int tid  = threadIdx.x;
    const int wave = tid >> 5;
    const int lane = tid & 31;
    const int l    = lane & 15;
    const int hi   = lane >> 4;

    const int m0 = blockIdx.y * 128;
    const int n0 = blockIdx.x * 128;
    const int wm = (wave >> 1) * 32;  // wave row offset in block tile
    const int wn = (wave & 1) * 64;   // wave col offset in block tile

    v8f acc[2][4];
#pragma unroll
    for (int i = 0; i < 2; ++i)
#pragma unroll
        for (int j = 0; j < 4; ++j)
            acc[i][j] = (v8f){0.f,0.f,0.f,0.f,0.f,0.f,0.f,0.f};

    const int sr = tid >> 1;   // staging row 0..127
    const int sh = tid & 1;    // which 16-elem half of the 32-wide k slice

    for (int k0 = 0; k0 < K; k0 += 32) {
        // stage 128x32 bf16 tiles of A and W into LDS (32B per thread per tile)
        const uint4* sa = (const uint4*)(A + (size_t)(m0 + sr) * K + k0 + sh * 16);
        uint4* da = (uint4*)(As + sr * 32 + sh * 16);
        da[0] = sa[0]; da[1] = sa[1];
        const uint4* sw = (const uint4*)(W + (size_t)(n0 + sr) * K + k0 + sh * 16);
        uint4* dw = (uint4*)(Ws + sr * 32 + sh * 16);
        dw[0] = sw[0]; dw[1] = sw[1];
        if (k0 + 32 < K) {  // hide next tile's L2/HBM latency behind the WMMA chain
            __builtin_prefetch(A + (size_t)(m0 + sr) * K + k0 + 32 + sh * 16, 0, 1);
            __builtin_prefetch(W + (size_t)(n0 + sr) * K + k0 + 32 + sh * 16, 0, 1);
        }
        __syncthreads();

        Frag fa[2], fb[4];
#pragma unroll
        for (int i = 0; i < 2; ++i) {
            // 16-bit A layout: lane<16 -> K{0..7,16..23}, lane>=16 -> K{8..15,24..31}
            const u16* p = As + (wm + 16 * i + l) * 32 + hi * 8;
            fa[i].q[0] = *(const uint4*)(p);
            fa[i].q[1] = *(const uint4*)(p + 16);
        }
#pragma unroll
        for (int j = 0; j < 4; ++j) {
            // 16-bit B layout: lane<16 -> K{0..7,16..23}, lane>=16 -> K{8..15,24..31} (col = W row)
            const u16* p = Ws + (wn + 16 * j + l) * 32 + hi * 8;
            fb[j].q[0] = *(const uint4*)(p);
            fb[j].q[1] = *(const uint4*)(p + 16);
        }
#pragma unroll
        for (int i = 0; i < 2; ++i)
#pragma unroll
            for (int j = 0; j < 4; ++j)
                acc[i][j] = __builtin_amdgcn_wmma_f32_16x16x32_bf16(
                    false, fa[i].bf, false, fb[j].bf, (short)0, acc[i][j], false, false);
        __syncthreads();
    }

#pragma unroll
    for (int i = 0; i < 2; ++i)
#pragma unroll
        for (int j = 0; j < 4; ++j)
#pragma unroll
            for (int rr = 0; rr < 8; ++rr) {
                int mi = m0 + wm + 16 * i + rr + 8 * hi;  // C layout: VGPR rr -> row rr (+8 for hi half)
                int ni = n0 + wn + 16 * j + l;
                float v = acc[i][j][rr] + bias[ni];
                if (writeF32) outF[(size_t)mi * N + ni] = v;
                else          outB[(size_t)mi * N + ni] = f2bf(v);
            }
}

// ---------------- attention: scores + softmax + PV, one WG per (b, 16-query tile) --------
__global__ __launch_bounds__(256) void attn_wmma(
    const u16* __restrict__ qb, const u16* __restrict__ kb,
    const u16* __restrict__ vtb,    // V transposed: [B,H,S] bf16
    const int* __restrict__ amask,
    float* __restrict__ attn_out,   // [B,1,S,S] fp32
    u16* __restrict__ ctxb,         // [B*S,H] bf16
    float scale)
{
    __shared__ __align__(16) u16 Psh[16 * Sc];  // 64KB score/prob tile (bf16)
    __shared__ float stats[256];                // [0..127] row max per wave, [128..255] row sum per wave

    const int tid  = threadIdx.x;
    const int wave = tid >> 5;
    const int lane = tid & 31;
    const int l    = lane & 15;
    const int hi   = lane >> 4;

    const int b    = blockIdx.y;
    const int q0   = blockIdx.x * 16;
    const int qMax = q0 + 15;
    const size_t rowQ = (size_t)b * Sc + q0;

    const u16 NEGBF = f2bf(-1.0e30f);

    // ---------------- phase 1: scores = scale * q @ k^T, masked; per-wave row max -------
    float rm[8];
#pragma unroll
    for (int r = 0; r < 8; ++r) rm[r] = -3.0e38f;

    for (int t = 0; t < 16; ++t) {
        const int keyBase = (wave * 16 + t) * 16;
        if (keyBase > qMax) {   // fully above the causal diagonal
#pragma unroll
            for (int r = 0; r < 8; ++r)
                Psh[(r + 8 * hi) * Sc + keyBase + l] = NEGBF;
            continue;
        }
        v8f acc = (v8f){0.f,0.f,0.f,0.f,0.f,0.f,0.f,0.f};
        const size_t rowK = (size_t)b * Sc + keyBase;
        for (int k0 = 0; k0 < Hc; k0 += 32) {
            Frag a, bb;
            const u16* ap = qb + (rowQ + l) * Hc + k0 + hi * 8;
            a.q[0] = *(const uint4*)(ap);
            a.q[1] = *(const uint4*)(ap + 16);
            const u16* bp = kb + (rowK + l) * Hc + k0 + hi * 8;
            bb.q[0] = *(const uint4*)(bp);
            bb.q[1] = *(const uint4*)(bp + 16);
            acc = __builtin_amdgcn_wmma_f32_16x16x32_bf16(
                false, a.bf, false, bb.bf, (short)0, acc, false, false);
        }
#pragma unroll
        for (int r = 0; r < 8; ++r) {
            const int m  = r + 8 * hi;
            const int qi = q0 + m;
            const int kj = keyBase + l;
            float s = acc[r] * scale;
            if (kj > qi || amask[(size_t)b * Sc + kj] == 0) s = -1.0e30f;
            rm[r] = fmaxf(rm[r], s);
            Psh[m * Sc + kj] = f2bf(s);
        }
    }
    // row-max reduce within each 16-lane half-wave, publish to LDS stats
#pragma unroll
    for (int r = 0; r < 8; ++r) {
        float v = rm[r];
        for (int off = 1; off < 16; off <<= 1) v = fmaxf(v, __shfl_xor(v, off, 16));
        if (l == 0) stats[wave * 16 + r + 8 * hi] = v;
    }
    __syncthreads();

    // ---------------- phase 2: p = exp(s - rowmax), per-wave row sums ----------------
    float fm[16];
#pragma unroll
    for (int m = 0; m < 16; ++m) {
        float v = stats[m];
        for (int w = 1; w < 8; ++w) v = fmaxf(v, stats[w * 16 + m]);
        fm[m] = v;
    }
    const int cs = wave * 256;  // this wave's key-column range
    for (int m = 0; m < 16; ++m) {
        float ps = 0.f;
#pragma unroll
        for (int j = 0; j < 8; ++j) {
            const int idx = m * Sc + cs + lane + 32 * j;
            float p = __expf(bf2f(Psh[idx]) - fm[m]);
            Psh[idx] = f2bf(p);
            ps += p;
        }
        for (int off = 1; off < 32; off <<= 1) ps += __shfl_xor(ps, off, 32);
        if (lane == 0) stats[128 + wave * 16 + m] = ps;
    }
    __syncthreads();

    // per-row 1/sum
    float invl[16];
#pragma unroll
    for (int m = 0; m < 16; ++m) {
        float sum = 0.f;
        for (int w = 0; w < 8; ++w) sum += stats[128 + w * 16 + m];
        invl[m] = 1.0f / sum;
    }

    // write normalized attention probabilities (fp32) for this wave's column range
    for (int m = 0; m < 16; ++m) {
        const size_t orow = ((size_t)b * Sc + q0 + m) * Sc;
#pragma unroll
        for (int j = 0; j < 8; ++j) {
            const int col = cs + lane + 32 * j;
            attn_out[orow + col] = bf2f(Psh[m * Sc + col]) * invl[m];
        }
    }

    // ---------------- phase 3: ctx = softmax(P) @ V  (A from LDS, B from V^T, contiguous) --
    for (int i = 0; i < 8; ++i) {
        const int n0 = (wave * 8 + i) * 16;
        // B-fragment row base: V^T row (b, n0+l) holds V[:, n0+l] contiguously over keys
        const u16* vrow = vtb + ((size_t)b * Hc + n0 + l) * Sc;
        v8f acc = (v8f){0.f,0.f,0.f,0.f,0.f,0.f,0.f,0.f};
        for (int c0 = 0; c0 <= qMax; c0 += 32) {   // causal: chunks past qMax are all zero
            Frag a, bb;
            const u16* ap = Psh + l * Sc + c0 + hi * 8;
            a.q[0] = *(const uint4*)(ap);
            a.q[1] = *(const uint4*)(ap + 16);
            const u16* bp = vrow + c0 + hi * 8;   // K=keys contiguous in V^T
            bb.q[0] = *(const uint4*)(bp);
            bb.q[1] = *(const uint4*)(bp + 16);
            acc = __builtin_amdgcn_wmma_f32_16x16x32_bf16(
                false, a.bf, false, bb.bf, (short)0, acc, false, false);
        }
#pragma unroll
        for (int r = 0; r < 8; ++r) {
            const int m = r + 8 * hi;
            ctxb[((size_t)b * Sc + q0 + m) * Hc + n0 + l] = f2bf(acc[r] * invl[m]);
        }
    }
}

// ---------------- host-side launcher ----------------
extern "C" void kernel_launch(void* const* d_in, const int* in_sizes, int n_in,
                              void* d_out, int out_size, void* d_ws, size_t ws_size,
                              hipStream_t stream)
{
    const float* x  = (const float*)d_in[0];
    const int*   am = (const int*)d_in[1];
    const float* Wq = (const float*)d_in[2];
    const float* bq = (const float*)d_in[3];
    const float* Wk = (const float*)d_in[4];
    const float* bk = (const float*)d_in[5];
    const float* Wv = (const float*)d_in[6];
    const float* bv = (const float*)d_in[7];
    const float* Wo = (const float*)d_in[8];
    const float* bo = (const float*)d_in[9];

    const size_t XN = (size_t)BSc * Hc;  // 8,388,608 elements
    const size_t WN = (size_t)Hc * Hc;   // 1,048,576 elements

    char* ws = (char*)d_ws;
    u16* xb   = (u16*)ws; ws += XN * 2;
    u16* wqb  = (u16*)ws; ws += WN * 2;
    u16* wkb  = (u16*)ws; ws += WN * 2;
    u16* wvb  = (u16*)ws; ws += WN * 2;
    u16* wob  = (u16*)ws; ws += WN * 2;
    u16* qb   = (u16*)ws; ws += XN * 2;
    u16* kb   = (u16*)ws; ws += XN * 2;
    u16* vb   = (u16*)ws; ws += XN * 2;
    u16* vtb  = (u16*)ws; ws += XN * 2;   // V transposed [B,H,S]
    u16* ctxb = (u16*)ws; ws += XN * 2;

    float* outp  = (float*)d_out;   // [B,S,H]
    float* attnp = outp + XN;       // [B,1,S,S]

    // fp32 -> bf16
    cvt_f32_bf16<<<4096, 256, 0, stream>>>(x,  xb,  (int)XN);
    cvt_f32_bf16<<<1024, 256, 0, stream>>>(Wq, wqb, (int)WN);
    cvt_f32_bf16<<<1024, 256, 0, stream>>>(Wk, wkb, (int)WN);
    cvt_f32_bf16<<<1024, 256, 0, stream>>>(Wv, wvb, (int)WN);
    cvt_f32_bf16<<<1024, 256, 0, stream>>>(Wo, wob, (int)WN);

    // Q/K/V projections (bf16 out, fp32 accumulate)
    dim3 gg(Hc / 128, BSc / 128);  // (8, 64)
    gemm_bf16_wmma<<<gg, 256, 0, stream>>>(xb, wqb, bq, qb, nullptr, BSc, Hc, Hc, 0);
    gemm_bf16_wmma<<<gg, 256, 0, stream>>>(xb, wkb, bk, kb, nullptr, BSc, Hc, Hc, 0);
    gemm_bf16_wmma<<<gg, 256, 0, stream>>>(xb, wvb, bv, vb, nullptr, BSc, Hc, Hc, 0);

    // V -> V^T (bf16), so PV B-fragments are contiguous loads
    dim3 tb(32, 8, 1);
    dim3 tg(Hc / 32, Sc / 32, Bc);  // (32, 64, 4)
    transpose_bf16<<<tg, tb, 0, stream>>>(vb, vtb);

    // attention (scores + softmax + PV), scale = 1/sqrt(H) = 1/32
    dim3 ga(Sc / 16, Bc);  // (128, 4)
    attn_wmma<<<ga, 256, 0, stream>>>(qb, kb, vtb, am, attnp, ctxb, 0.03125f);

    // output projection (fp32 out)
    gemm_bf16_wmma<<<gg, 256, 0, stream>>>(ctxb, wob, bo, nullptr, outp, BSc, Hc, Hc, 1);
}
